// TransformerEncoderLayer_7241314861802
// MI455X (gfx1250) — matrix operand order repaired and checked by reading, not compiled, not run on hardware
//
#include <hip/hip_runtime.h>
#include <hip/hip_bf16.h>

// ---------------- constants ----------------
#define BATCH   4
#define SEQ     2048
#define HID     512
#define NHEAD   8
#define HDIM    64
#define INNER   2048
#define ROWS    (BATCH * SEQ)          // 8192
#define MAXD    512

typedef __attribute__((ext_vector_type(16))) __bf16 v16bf;
typedef __attribute__((ext_vector_type(8)))  __bf16 v8bf;
typedef __attribute__((ext_vector_type(8)))  float  v8f;
typedef __attribute__((ext_vector_type(4)))  int    v4i;

#define AS1 __attribute__((address_space(1)))
#define AS3 __attribute__((address_space(3)))

// CDNA5 async global->LDS path (guarded: falls back to load+ds_store if absent)
#if __has_builtin(__builtin_amdgcn_global_load_async_to_lds_b128)
#define HAVE_ASYNC_LDS 1
#else
#define HAVE_ASYNC_LDS 0
#endif

#if HAVE_ASYNC_LDS
__device__ __forceinline__ void async_ld16(const void* g, void* l) {
  __builtin_amdgcn_global_load_async_to_lds_b128((AS1 v4i*)g, (AS3 v4i*)l, 0, 0);
}
#if __has_builtin(__builtin_amdgcn_s_wait_asynccnt)
#define WAIT_ASYNC(n) __builtin_amdgcn_s_wait_asynccnt(n)
#else
#define WAIT_ASYNC(n) asm volatile("s_wait_asynccnt %0" ::"n"(n) : "memory")
#endif
#endif

__device__ __forceinline__ __bf16 f2bf(float f) {
  union { float f; unsigned u; } v; v.f = f;
  unsigned r = v.u + 0x7FFFu + ((v.u >> 16) & 1u);   // round-to-nearest-even
  unsigned short h = (unsigned short)(r >> 16);
  return __builtin_bit_cast(__bf16, h);
}

__device__ __forceinline__ v8f zero8() {
  v8f z;
#pragma unroll
  for (int i = 0; i < 8; ++i) z[i] = 0.0f;
  return z;
}

// Load a 16-element bf16 fragment from two 16-byte chunks.
__device__ __forceinline__ v16bf load_frag2(const __bf16* p0, const __bf16* p1) {
  v8bf lo = *(const v8bf*)p0;
  v8bf hi = *(const v8bf*)p1;
  v16bf r;
#pragma unroll
  for (int i = 0; i < 8; ++i) { r[i] = lo[i]; r[i + 8] = hi[i]; }
  return r;
}

// A-matrix (16x32, row = lane&15): element e -> K = half*16+e
__device__ __forceinline__ v16bf load_afrag(const __bf16* rowp, int half) {
  return load_frag2(rowp + half * 16, rowp + half * 16 + 8);
}
// B-matrix (32x16, col = lane&15, stored as B^T row): element e -> K = half*16+e
__device__ __forceinline__ v16bf load_bfrag(const __bf16* rowp, int half) {
  return load_frag2(rowp + half * 16, rowp + half * 16 + 8);
}

#define WMMA_BF16(a, b, c) \
  __builtin_amdgcn_wmma_f32_16x16x32_bf16(false, (a), false, (b), (short)0, (c), false, false)

// ---------------- elementwise converts ----------------
__global__ void cvt_f32_to_bf16(const float* __restrict__ in, __bf16* __restrict__ out, int n) {
  int i = blockIdx.x * blockDim.x + threadIdx.x;
  if (i < n) out[i] = f2bf(in[i]);
}

// W[K,N] fp32 -> Wt[N,K] bf16
__global__ void cvt_transpose(const float* __restrict__ W, __bf16* __restrict__ Wt, int K, int N) {
  int i = blockIdx.x * blockDim.x + threadIdx.x;
  if (i >= K * N) return;
  int k = i / N, n = i - k * N;
  Wt[(size_t)n * K + k] = f2bf(W[i]);
}

// ---------------- WMMA GEMM with async-LDS double-buffered B staging ----------------
// C[M,N] = A[M,K](bf16,rm) @ Wt[N,K]^T (bf16) + bias[N]
// block: 256 thr (8 waves). Block tile: 128 M x 64 N. Wave w owns m-rows [by*128+w*16, +16).
// The 64x32 B-tile (4KB) is shared by all 8 waves, staged in LDS, double-buffered:
// stage i+1 fetched (async global->LDS) while stage i feeds the WMMAs. Last step peeled
// so the steady-state loop is branch-free.
__global__ __launch_bounds__(256)
void gemm_wmma(const __bf16* __restrict__ A, const __bf16* __restrict__ Wt,
               const float* __restrict__ bias,
               float* __restrict__ Cf, __bf16* __restrict__ Cb,
               int M, int N, int K, int mode,
               __bf16* __restrict__ qo, __bf16* __restrict__ ko,
               __bf16* __restrict__ vto) {
  __shared__ __bf16 btile[2][64 * 32];
  const int tid  = threadIdx.x;
  const int lane = tid & 31;
  const int wave = tid >> 5;
  const int n0 = blockIdx.x * 64;
  const int m0 = blockIdx.y * 128 + wave * 16;
  const int row  = lane & 15;
  const int half = lane >> 4;

  // staging map: thread t copies 16B: B row n0+(t>>2), k-chunk (t&3)*8 elements
  const int ldrow = tid >> 2, ldchunk = (tid & 3) * 8;
  const __bf16* bsrc = Wt + (size_t)(n0 + ldrow) * K + ldchunk;
  __bf16* bdst0 = &btile[0][ldrow * 32 + ldchunk];
  __bf16* bdst1 = &btile[1][ldrow * 32 + ldchunk];

  v8f acc[4];
#pragma unroll
  for (int j = 0; j < 4; ++j) acc[j] = zero8();

  const __bf16* arow = A + (size_t)(m0 + row) * K;

  // hoist all 4 B fragments before the 4 WMMAs so DS latency overlaps the MACs
  auto compute = [&](v16bf af, int bufn) {
    const __bf16* bbase = &btile[bufn][half * 16];
    const __bf16* p0 = bbase + row * 32;
    const __bf16* p1 = bbase + (16 + row) * 32;
    const __bf16* p2 = bbase + (32 + row) * 32;
    const __bf16* p3 = bbase + (48 + row) * 32;
    v16bf b0 = load_frag2(p0, p0 + 8);
    v16bf b1 = load_frag2(p1, p1 + 8);
    v16bf b2 = load_frag2(p2, p2 + 8);
    v16bf b3 = load_frag2(p3, p3 + 8);
    acc[0] = WMMA_BF16(af, b0, acc[0]);
    acc[1] = WMMA_BF16(af, b1, acc[1]);
    acc[2] = WMMA_BF16(af, b2, acc[2]);
    acc[3] = WMMA_BF16(af, b3, acc[3]);
  };

  int k0 = 0;
#if HAVE_ASYNC_LDS
  async_ld16(bsrc, bdst0);                       // prologue: stage 0 -> buf0
  int buf = 0;
  for (; k0 < K - 32; k0 += 32) {                // branch-free steady state
    async_ld16(bsrc + k0 + 32, buf ? bdst0 : bdst1);
    v16bf af = load_afrag(arow + k0, half);
    __builtin_prefetch(arow + k0 + 64, 0, 3);
    WAIT_ASYNC(1);
    __syncthreads();
    compute(af, buf);
    __syncthreads();
    buf ^= 1;
  }
  {                                              // peeled last step
    v16bf af = load_afrag(arow + k0, half);
    WAIT_ASYNC(0);
    __syncthreads();
    compute(af, buf);
  }
#else
  for (; k0 < K; k0 += 32) {
    *(v8bf*)bdst0 = *(const v8bf*)(bsrc + k0);
    v16bf af = load_afrag(arow + k0, half);
    __syncthreads();
    compute(af, 0);
    __syncthreads();
  }
#endif

#pragma unroll
  for (int j = 0; j < 4; ++j) {
    const int n = n0 + j * 16 + row;
    const float bv = bias ? bias[n] : 0.0f;
#pragma unroll
    for (int r = 0; r < 8; ++r) {
      const int m = m0 + r + 8 * half;
      float v = acc[j][r] + bv;
      if (mode == 1) v = 0.5f * v * (1.0f + erff(v * 0.70710678118f));
      if (mode == 2) {
        const int t = m & (SEQ - 1);
        const int b = m >> 11;                 // SEQ = 2048
        const int which = n >> 9;              // 0=q 1=k 2=v
        const int h = (n >> 6) & (NHEAD - 1);
        const int d = n & (HDIM - 1);
        const size_t bh = (size_t)(b * NHEAD + h);
        if (which == 0)       qo[(bh * SEQ + t) * HDIM + d] = f2bf(v);
        else if (which == 1)  ko[(bh * SEQ + t) * HDIM + d] = f2bf(v);
        else                  vto[(bh * HDIM + d) * SEQ + t] = f2bf(v);
      } else {
        const size_t o = (size_t)m * N + n;
        if (Cf) Cf[o] = v;
        if (Cb) Cb[o] = f2bf(v);
      }
    }
  }
}

// ---------------- flash attention with WMMA + async-staged shared K/V ----------------
// grid: (SEQ/64, NHEAD, BATCH), block 128 (4 waves; wave w owns q-tile bx*64+w*16).
// All 4 waves share one (b,h): the 32-key K-chunk (32x64, 4KB) and V^T-chunk (64x32, 4KB)
// are staged in LDS via async global->LDS, double-buffered across the key loop.
__global__ __launch_bounds__(128)
void attn_flash(const __bf16* __restrict__ Q, const __bf16* __restrict__ Kb,
                const __bf16* __restrict__ Vt, const float* __restrict__ rel,
                float* __restrict__ O) {
  __shared__ __bf16 ktile[2][32 * 64];
  __shared__ __bf16 vtile[2][64 * 32];
  __shared__ float  p_lds[4][16 * 33];
  const int tid  = threadIdx.x;
  const int lane = tid & 31;
  const int wave = tid >> 5;
  const int q0 = blockIdx.x * 64 + wave * 16;
  const int h  = blockIdx.y;
  const int b  = blockIdx.z;
  const int row  = lane & 15;
  const int half = lane >> 4;
  const size_t bh = (size_t)(b * NHEAD + h);

  // per-thread staging map (4 async b128 per 32-key chunk):
  // K chunk: 32 rows x 64 elems; thread stages rows tid>>3 and 16+(tid>>3), 16B at (tid&7)*8
  const int krow0 = tid >> 3, koff = (tid & 7) * 8;
  const __bf16* ksrc0 = Kb + (bh * SEQ + krow0) * HDIM + koff;
  const __bf16* ksrc1 = ksrc0 + 16 * HDIM;
  __bf16* kdst0[2] = { &ktile[0][krow0 * 64 + koff], &ktile[1][krow0 * 64 + koff] };
  __bf16* kdst1[2] = { &ktile[0][(16 + krow0) * 64 + koff], &ktile[1][(16 + krow0) * 64 + koff] };
  // V chunk: 64 rows x 32 elems; thread stages rows tid>>2 and 32+(tid>>2), 16B at (tid&3)*8
  const int vrow0 = tid >> 2, voff = (tid & 3) * 8;
  const __bf16* vsrc0 = Vt + (bh * HDIM + vrow0) * SEQ + voff;
  const __bf16* vsrc1 = vsrc0 + (size_t)32 * SEQ;
  __bf16* vdst0[2] = { &vtile[0][vrow0 * 32 + voff], &vtile[1][vrow0 * 32 + voff] };
  __bf16* vdst1[2] = { &vtile[0][(32 + vrow0) * 32 + voff], &vtile[1][(32 + vrow0) * 32 + voff] };

  const __bf16* qrow = Q + (bh * SEQ + q0 + row) * HDIM;
  const v16bf qf0 = load_afrag(qrow, half);
  const v16bf qf1 = load_afrag(qrow + 32, half);

  float m_i[8], l_i[8];
  v8f o[4];
#pragma unroll
  for (int r = 0; r < 8; ++r) { m_i[r] = -1e30f; l_i[r] = 0.0f; }
#pragma unroll
  for (int j = 0; j < 4; ++j) o[j] = zero8();

#if HAVE_ASYNC_LDS
  auto stage = [&](int kcs, int bufn) {
    async_ld16(ksrc0 + (size_t)kcs * HDIM, kdst0[bufn]);
    async_ld16(ksrc1 + (size_t)kcs * HDIM, kdst1[bufn]);
    async_ld16(vsrc0 + kcs, vdst0[bufn]);
    async_ld16(vsrc1 + kcs, vdst1[bufn]);
  };
#else
  auto stage_sync = [&](int kcs, int bufn) {
    *(v8bf*)kdst0[bufn] = *(const v8bf*)(ksrc0 + (size_t)kcs * HDIM);
    *(v8bf*)kdst1[bufn] = *(const v8bf*)(ksrc1 + (size_t)kcs * HDIM);
    *(v8bf*)vdst0[bufn] = *(const v8bf*)(vsrc0 + kcs);
    *(v8bf*)vdst1[bufn] = *(const v8bf*)(vsrc1 + kcs);
  };
#endif

  // process one 32-key chunk out of LDS buffer bufn
  auto process = [&](int kc, int bufn) {
    const __bf16* kt = &ktile[bufn][0];
    const __bf16* vt = &vtile[bufn][0];
    // S = Q K^T (two 16-col tiles, K-dim 64 = 2 WMMA each)
    const __bf16* kb0 = kt + row * 64 + half * 16;
    const __bf16* kb1 = kt + (16 + row) * 64 + half * 16;
    v16bf kf00 = load_frag2(kb0,      kb0 + 8);
    v16bf kf01 = load_frag2(kb0 + 32, kb0 + 40);
    v16bf kf10 = load_frag2(kb1,      kb1 + 8);
    v16bf kf11 = load_frag2(kb1 + 32, kb1 + 40);
    v8f s0 = zero8(), s1 = zero8();
    s0 = WMMA_BF16(qf0, kf00, s0);
    s0 = WMMA_BF16(qf1, kf01, s0);
    s1 = WMMA_BF16(qf0, kf10, s1);
    s1 = WMMA_BF16(qf1, kf11, s1);

    // scale + rel-pos bias + online softmax
    float p0[8], p1[8], alpha[8];
#pragma unroll
    for (int r = 0; r < 8; ++r) {
      const int qp  = q0 + r + 8 * half;
      const int kp0 = kc + row;
      const int kp1 = kc + 16 + row;
      int d0 = qp - kp0; d0 = d0 < -MAXD ? -MAXD : (d0 > MAXD ? MAXD : d0);
      int d1 = qp - kp1; d1 = d1 < -MAXD ? -MAXD : (d1 > MAXD ? MAXD : d1);
      float v0 = s0[r] * 0.125f + rel[(d0 + MAXD) * NHEAD + h];
      float v1 = s1[r] * 0.125f + rel[(d1 + MAXD) * NHEAD + h];
      float cm = fmaxf(v0, v1);
      cm = fmaxf(cm, __shfl_xor(cm, 1));
      cm = fmaxf(cm, __shfl_xor(cm, 2));
      cm = fmaxf(cm, __shfl_xor(cm, 4));
      cm = fmaxf(cm, __shfl_xor(cm, 8));
      const float mn = fmaxf(m_i[r], cm);
      alpha[r] = __expf(m_i[r] - mn);
      p0[r] = __expf(v0 - mn);
      p1[r] = __expf(v1 - mn);
      float rs = p0[r] + p1[r];
      rs += __shfl_xor(rs, 1);
      rs += __shfl_xor(rs, 2);
      rs += __shfl_xor(rs, 4);
      rs += __shfl_xor(rs, 8);
      l_i[r] = l_i[r] * alpha[r] + rs;
      m_i[r] = mn;
    }
#pragma unroll
    for (int j = 0; j < 4; ++j)
#pragma unroll
      for (int r = 0; r < 8; ++r) o[j][r] *= alpha[r];

    // transpose P (C/D layout -> A layout) through this wave's private LDS tile
    float* pw = &p_lds[wave][0];
#pragma unroll
    for (int r = 0; r < 8; ++r) {
      pw[(r + 8 * half) * 33 + row]      = p0[r];
      pw[(r + 8 * half) * 33 + 16 + row] = p1[r];
    }
    __syncthreads();
    v16bf pf;
#pragma unroll
    for (int e = 0; e < 16; ++e) {
      const int kk = half * 16 + e;
      pf[e] = f2bf(pw[row * 33 + kk]);
    }

    // O += P @ V
#pragma unroll
    for (int j = 0; j < 4; ++j) {
      const __bf16* vp = vt + (j * 16 + row) * 32 + half * 16;
      v16bf vf = load_frag2(vp, vp + 8);
      o[j] = WMMA_BF16(pf, vf, o[j]);
    }
  };

#if HAVE_ASYNC_LDS
  stage(0, 0);
  int buf = 0, kc = 0;
  for (; kc < SEQ - 32; kc += 32) {              // branch-free steady state
    stage(kc + 32, buf ^ 1);
    WAIT_ASYNC(4);
    __syncthreads();
    process(kc, buf);
    __syncthreads();
    buf ^= 1;
  }
  WAIT_ASYNC(0);
  __syncthreads();
  process(kc, buf);
#else
  for (int kc = 0; kc < SEQ; kc += 32) {
    stage_sync(kc, 0);
    __syncthreads();
    process(kc, 0);
    __syncthreads();
  }
#endif

#pragma unroll
  for (int j = 0; j < 4; ++j) {
    const int d = j * 16 + row;
#pragma unroll
    for (int r = 0; r < 8; ++r) {
      const int qp = q0 + r + 8 * half;
      O[(bh * SEQ + qp) * HDIM + d] = o[j][r] / l_i[r];
    }
  }
}

// [B,H,T,D] fp32 -> [B*T, C] bf16
__global__ void permute_attn(const float* __restrict__ attn, __bf16* __restrict__ out) {
  const size_t i = (size_t)blockIdx.x * blockDim.x + threadIdx.x;
  if (i >= (size_t)BATCH * NHEAD * SEQ * HDIM) return;
  const int d = (int)(i & (HDIM - 1));
  const size_t r = i >> 6;
  const int t = (int)(r & (SEQ - 1));
  const size_t bhh = r >> 11;
  const int h = (int)(bhh & (NHEAD - 1));
  const int b = (int)(bhh >> 3);
  out[((size_t)(b * SEQ + t)) * HID + h * HDIM + d] = f2bf(attn[i]);
}

// out = LayerNorm(X + Y) * g + b ; one block (256 thr) per row of 512
__global__ void residual_ln(const float* __restrict__ X, const float* __restrict__ Y,
                            const float* __restrict__ g, const float* __restrict__ bb,
                            float* __restrict__ outf, __bf16* __restrict__ outb) {
  const int rowi = blockIdx.x;
  const int tid = threadIdx.x;
  const float* xr = X + (size_t)rowi * HID;
  const float* yr = Y + (size_t)rowi * HID;
  const float v0 = xr[tid] + yr[tid];
  const float v1 = xr[tid + 256] + yr[tid + 256];
  float s = v0 + v1;
  float ss = v0 * v0 + v1 * v1;
#pragma unroll
  for (int mask = 16; mask >= 1; mask >>= 1) {
    s  += __shfl_xor(s, mask);
    ss += __shfl_xor(ss, mask);
  }
  __shared__ float shs[8], shss[8];
  __shared__ float mu, rstd;
  if ((tid & 31) == 0) { shs[tid >> 5] = s; shss[tid >> 5] = ss; }
  __syncthreads();
  if (tid == 0) {
    float S = 0.0f, SS = 0.0f;
#pragma unroll
    for (int i = 0; i < 8; ++i) { S += shs[i]; SS += shss[i]; }
    const float m = S * (1.0f / HID);
    const float var = SS * (1.0f / HID) - m * m;
    mu = m;
    rstd = rsqrtf(var + 1e-5f);
  }
  __syncthreads();
  const float o0 = (v0 - mu) * rstd * g[tid] + bb[tid];
  const float o1 = (v1 - mu) * rstd * g[tid + 256] + bb[tid + 256];
  const size_t base = (size_t)rowi * HID;
  if (outf) { outf[base + tid] = o0; outf[base + tid + 256] = o1; }
  if (outb) { outb[base + tid] = f2bf(o0); outb[base + tid + 256] = f2bf(o1); }
}

// ---------------- host-side launch ----------------
extern "C" void kernel_launch(void* const* d_in, const int* in_sizes, int n_in,
                              void* d_out, int out_size, void* d_ws, size_t ws_size,
                              hipStream_t stream) {
  (void)in_sizes; (void)n_in; (void)out_size; (void)ws_size;
  const float* x       = (const float*)d_in[0];
  const float* w_qkv   = (const float*)d_in[1];
  const float* b_qkv   = (const float*)d_in[2];
  const float* w_out   = (const float*)d_in[3];
  const float* b_out   = (const float*)d_in[4];
  const float* ln1_g   = (const float*)d_in[5];
  const float* ln1_b   = (const float*)d_in[6];
  const float* ln2_g   = (const float*)d_in[7];
  const float* ln2_b   = (const float*)d_in[8];
  const float* w_mlp1  = (const float*)d_in[9];
  const float* b_mlp1  = (const float*)d_in[10];
  const float* w_mlp2  = (const float*)d_in[11];
  const float* b_mlp2  = (const float*)d_in[12];
  const float* rel     = (const float*)d_in[13];
  float* out = (float*)d_out;

  char* ws = (char*)d_ws;
  size_t off = 0;
  auto alloc = [&](size_t bytes) { size_t c = off; off += (bytes + 255) & ~(size_t)255; return c; };
  const size_t XB    = alloc((size_t)ROWS * HID * 2);           // x bf16
  const size_t WQKVT = alloc((size_t)(3 * HID) * HID * 2);      // [1536,512] bf16
  const size_t WOUTT = alloc((size_t)HID * HID * 2);            // [512,512]  bf16
  const size_t WM1T  = alloc((size_t)INNER * HID * 2);          // [2048,512] bf16
  const size_t WM2T  = alloc((size_t)HID * INNER * 2);          // [512,2048] bf16
  const size_t QB    = alloc((size_t)ROWS * HID * 2);           // q  [B,H,T,D] bf16
  const size_t KB    = alloc((size_t)ROWS * HID * 2);           // k  [B,H,T,D] bf16
  const size_t VTB   = alloc((size_t)ROWS * HID * 2);           // vT [B,H,D,T] bf16
  const size_t AT32  = alloc((size_t)ROWS * HID * 4);           // attn fp32 / proj fp32 (aliased)
  const size_t ATC   = alloc((size_t)ROWS * HID * 2);           // attn ctx bf16 [B*T,C]
  const size_t X1    = alloc((size_t)ROWS * HID * 4);           // LN1 out fp32
  const size_t X1B   = alloc((size_t)ROWS * HID * 2);           // LN1 out bf16
  const size_t HB    = alloc((size_t)ROWS * INNER * 2);         // gelu(h) bf16
  const size_t M2F   = QB;                                      // mlp2 fp32 reuses dead q/k region

  __bf16* xb    = (__bf16*)(ws + XB);
  __bf16* wqkvt = (__bf16*)(ws + WQKVT);
  __bf16* woutt = (__bf16*)(ws + WOUTT);
  __bf16* wm1t  = (__bf16*)(ws + WM1T);
  __bf16* wm2t  = (__bf16*)(ws + WM2T);
  __bf16* qb    = (__bf16*)(ws + QB);
  __bf16* kb    = (__bf16*)(ws + KB);
  __bf16* vtb   = (__bf16*)(ws + VTB);
  float*  attnf = (float*)(ws + AT32);
  float*  projf = (float*)(ws + AT32);
  __bf16* attnc = (__bf16*)(ws + ATC);
  float*  x1    = (float*)(ws + X1);
  __bf16* x1b   = (__bf16*)(ws + X1B);
  __bf16* hb    = (__bf16*)(ws + HB);
  float*  m2f   = (float*)(ws + M2F);

  {
    int n = ROWS * HID;
    cvt_f32_to_bf16<<<(n + 255) / 256, 256, 0, stream>>>(x, xb, n);
  }
  cvt_transpose<<<(HID * 3 * HID + 255) / 256, 256, 0, stream>>>(w_qkv, wqkvt, HID, 3 * HID);
  cvt_transpose<<<(HID * HID + 255) / 256, 256, 0, stream>>>(w_out, woutt, HID, HID);
  cvt_transpose<<<(HID * INNER + 255) / 256, 256, 0, stream>>>(w_mlp1, wm1t, HID, INNER);
  cvt_transpose<<<(INNER * HID + 255) / 256, 256, 0, stream>>>(w_mlp2, wm2t, INNER, HID);

  // QKV GEMM with q/k/vT scatter epilogue
  gemm_wmma<<<dim3((3 * HID) / 64, ROWS / 128), 256, 0, stream>>>(
      xb, wqkvt, b_qkv, nullptr, nullptr, ROWS, 3 * HID, HID, 2, qb, kb, vtb);

  // flash attention (4 waves/block share async-staged K/V chunks)
  attn_flash<<<dim3(SEQ / 64, NHEAD, BATCH), 128, 0, stream>>>(qb, kb, vtb, rel, attnf);

  {
    int n = ROWS * HID;
    permute_attn<<<(n + 255) / 256, 256, 0, stream>>>(attnf, attnc);
  }

  // output projection
  gemm_wmma<<<dim3(HID / 64, ROWS / 128), 256, 0, stream>>>(
      attnc, woutt, b_out, projf, nullptr, ROWS, HID, HID, 0, nullptr, nullptr, nullptr);

  // x1 = LN1(x + proj)
  residual_ln<<<ROWS, 256, 0, stream>>>(x, projf, ln1_g, ln1_b, x1, x1b);

  // h = GELU(x1 @ w_mlp1 + b1)
  gemm_wmma<<<dim3(INNER / 64, ROWS / 128), 256, 0, stream>>>(
      x1b, wm1t, b_mlp1, nullptr, hb, ROWS, INNER, HID, 1, nullptr, nullptr, nullptr);

  // mlp2 = h @ w_mlp2 + b2
  gemm_wmma<<<dim3(HID / 64, ROWS / 128), 256, 0, stream>>>(
      hb, wm2t, b_mlp2, m2f, nullptr, ROWS, HID, INNER, 0, nullptr, nullptr, nullptr);

  // out = LN2(x1 + mlp2)
  residual_ln<<<ROWS, 256, 0, stream>>>(x1, m2f, ln2_g, ln2_b, out, nullptr);
}
